// GAT_layer_75754633167660
// MI455X (gfx1250) — matrix operand order repaired and checked by reading, not compiled, not run on hardware
//
#include <hip/hip_runtime.h>
#include <hip/hip_fp16.h>
#include <math.h>

typedef __attribute__((ext_vector_type(16))) _Float16 v16h;
typedef __attribute__((ext_vector_type(8)))  _Float16 v8h;
typedef __attribute__((ext_vector_type(8)))  float    v8f;

#define N_NODES 8192
#define F_IN    512
#define F_OUT   64
#define NCHUNK  4
#define CHUNK_J (N_NODES / NCHUNK)   // 2048 columns of A per wave

// ---------------------------------------------------------------------------
// Kernel 1: H = X @ W (fp32), emit HT (f16, transposed [F_OUT][N]) for WMMA
// B-operands, and f1 = H@a1, f2 = H@a2. One block (64 threads) per row.
// Memory-trivial: 16 MB of X total; W (128 KB) stays resident in L2.
// ---------------------------------------------------------------------------
__global__ void __launch_bounds__(64)
gat_hidden_kernel(const float* __restrict__ X, const float* __restrict__ W,
                  const float* __restrict__ a,
                  float* __restrict__ f1, float* __restrict__ f2,
                  _Float16* __restrict__ HT)
{
    __shared__ float Xs[F_IN];
    __shared__ float red1[2];   // per-wave partials, wave 0 / wave 1
    __shared__ float red2[2];
    const int i = blockIdx.x;
    const int t = threadIdx.x;

    for (int k = t; k < F_IN; k += 64) Xs[k] = X[(size_t)i * F_IN + k];
    __syncthreads();

    float acc = 0.f;
#pragma unroll 8
    for (int k = 0; k < F_IN; ++k)
        acc = fmaf(Xs[k], W[(size_t)k * F_OUT + t], acc);

    // f16 transposed copy for attention B-operands (feature-major, contiguous j)
    HT[(size_t)t * N_NODES + i] = (_Float16)acc;

    // wave-level tree reduce of acc*a1 / acc*a2 (wave32)
    float p1 = acc * a[t];
    float p2 = acc * a[F_OUT + t];
#pragma unroll
    for (int off = 16; off > 0; off >>= 1) {
        p1 += __shfl_xor(p1, off, 32);
        p2 += __shfl_xor(p2, off, 32);
    }
    if ((t & 31) == 0) { red1[t >> 5] = p1; red2[t >> 5] = p2; }
    __syncthreads();
    if (t == 0) f1[i] = red1[0] + red1[1];
    if (t == 1) f2[i] = red2[0] + red2[1];
}

// ---------------------------------------------------------------------------
// Kernel 2: fused masked-softmax-numerator + P@H via WMMA.
// 1 wave (32 threads) handles 16 rows x one j-chunk of 2048.
// Per j-step of 32: build P (16x32 f16, WMMA A layout), load 4 B tiles
// (32x16 f16 from HT), 4x v_wmma_f32_16x16x32_f16. Unrolled x2 so the
// scheduler interleaves two independent P/load/WMMA streams.
// A-operand element mapping (dense 16-bit A 16x32, ISA 7.12.2):
//   lane: h = lane>>4, M = lane&15; element e: K = (e>>3)*16 + h*8 + (e&7)
// B-operand (32x16, rows striped across lanes within a VGPR):
//   lane: N = lane&15, h = lane>>4; element e: K = (e>>3)*16 + h*8 + (e&7)
// C/D f32 16x16: lane N = lane&15; VGPR r holds row M = (lane>>4)*8 + r
// ---------------------------------------------------------------------------
__global__ void __launch_bounds__(32)
gat_attn_kernel(const int* __restrict__ A, const float* __restrict__ f1,
                const float* __restrict__ f2, const _Float16* __restrict__ HT,
                float* __restrict__ Spart, float* __restrict__ lpart)
{
    const int rb    = blockIdx.x >> 2;   // row block 0..511
    const int chunk = blockIdx.x & 3;    // j chunk 0..3
    const int lane  = threadIdx.x;
    const int h     = lane >> 4;
    const int M     = lane & 15;
    const int row   = rb * 16 + M;
    const float f1i = f1[row];

    v8f acc0 = {}, acc1 = {}, acc2 = {}, acc3 = {};
    float rowsum = 0.f;

    const size_t arow  = (size_t)row * N_NODES;
    const int    jbase = chunk * CHUNK_J;

#pragma unroll 2
    for (int it = 0; it < CHUNK_J / 32; ++it) {
        const int j0 = jbase + it * 32;

        // speculative WGP-scope prefetch of the mask stream two tiles ahead
        // (prefetch past the end of A is silently dropped)
        __builtin_prefetch(A + arow + j0 + 64, 0, 3);

        // ---- mask + f2 for this lane's 16 K positions (A-operand layout) ----
        const int4*   mp0 = (const int4*)  (A  + arow + j0 +      h * 8);
        const int4*   mp1 = (const int4*)  (A  + arow + j0 + 16 + h * 8);
        const float4* fp0 = (const float4*)(f2 + j0 +      h * 8);
        const float4* fp1 = (const float4*)(f2 + j0 + 16 + h * 8);
        int4   m0a = mp0[0], m0b = mp0[1], m1a = mp1[0], m1b = mp1[1];
        float4 g0a = fp0[0], g0b = fp0[1], g1a = fp1[0], g1b = fp1[1];

        auto pv = [&](int m, float fv) -> float {
            float s = f1i + fv;
            float e = s > 0.f ? s : 0.2f * s;       // leaky_relu(0.2)
            float r = (m > 0) ? __expf(e) : 0.f;    // masked exp (no max needed)
            rowsum += r;
            return r;
        };

        float p[16];
        p[ 0]=pv(m0a.x,g0a.x); p[ 1]=pv(m0a.y,g0a.y); p[ 2]=pv(m0a.z,g0a.z); p[ 3]=pv(m0a.w,g0a.w);
        p[ 4]=pv(m0b.x,g0b.x); p[ 5]=pv(m0b.y,g0b.y); p[ 6]=pv(m0b.z,g0b.z); p[ 7]=pv(m0b.w,g0b.w);
        p[ 8]=pv(m1a.x,g1a.x); p[ 9]=pv(m1a.y,g1a.y); p[10]=pv(m1a.z,g1a.z); p[11]=pv(m1a.w,g1a.w);
        p[12]=pv(m1b.x,g1b.x); p[13]=pv(m1b.y,g1b.y); p[14]=pv(m1b.z,g1b.z); p[15]=pv(m1b.w,g1b.w);

        v16h aop;
#pragma unroll
        for (int e = 0; e < 16; ++e) aop[e] = (_Float16)p[e];

        // ---- B tiles: HT[(n0+N)][j0 + (e>>3)*16 + h*8 + (e&7)], two runs of 8 f16 ----
        v16h b[4];
#pragma unroll
        for (int tN = 0; tN < 4; ++tN) {
            const _Float16* base = HT + (size_t)(tN * 16 + M) * N_NODES + j0 + h * 8;
            v8h lo = *(const v8h*)(base);
            v8h hi = *(const v8h*)(base + 16);
#pragma unroll
            for (int e = 0; e < 8; ++e) { b[tN][e] = lo[e]; b[tN][8 + e] = hi[e]; }
        }

        acc0 = __builtin_amdgcn_wmma_f32_16x16x32_f16(false, aop, false, b[0], (short)0, acc0, false, false);
        acc1 = __builtin_amdgcn_wmma_f32_16x16x32_f16(false, aop, false, b[1], (short)0, acc1, false, false);
        acc2 = __builtin_amdgcn_wmma_f32_16x16x32_f16(false, aop, false, b[2], (short)0, acc2, false, false);
        acc3 = __builtin_amdgcn_wmma_f32_16x16x32_f16(false, aop, false, b[3], (short)0, acc3, false, false);
    }

    // combine the two K-halves of each row's partial sum (lane <-> lane^16)
    rowsum += __shfl_xor(rowsum, 16, 32);
    if (lane < 16)
        lpart[(size_t)chunk * N_NODES + row] = rowsum;

    // store S partial: acc[t][r] = S[rb*16 + h*8 + r][t*16 + N]
    float* Sp = Spart + ((size_t)chunk * N_NODES + (size_t)rb * 16) * F_OUT;
    const int N = lane & 15;
#pragma unroll
    for (int r = 0; r < 8; ++r) {
        const size_t m = (size_t)(h * 8 + r) * F_OUT;
        Sp[m +  0 + N] = acc0[r];
        Sp[m + 16 + N] = acc1[r];
        Sp[m + 32 + N] = acc2[r];
        Sp[m + 48 + N] = acc3[r];
    }
}

// ---------------------------------------------------------------------------
// Kernel 3: combine j-chunk partials, normalize, ELU.
// ---------------------------------------------------------------------------
__global__ void __launch_bounds__(256)
gat_final_kernel(const float* __restrict__ Spart, const float* __restrict__ lpart,
                 float* __restrict__ out)
{
    const int idx = blockIdx.x * 256 + threadIdx.x;
    if (idx >= N_NODES * F_OUT) return;
    const int i = idx >> 6;   // row
    float s = 0.f, l = 0.f;
#pragma unroll
    for (int c = 0; c < NCHUNK; ++c) {
        s += Spart[(size_t)c * N_NODES * F_OUT + idx];
        l += lpart[(size_t)c * N_NODES + i];
    }
    const float hn = s / l;
    out[idx] = hn > 0.f ? hn : (__expf(hn) - 1.f);   // ELU
}

// ---------------------------------------------------------------------------
extern "C" void kernel_launch(void* const* d_in, const int* in_sizes, int n_in,
                              void* d_out, int out_size, void* d_ws, size_t ws_size,
                              hipStream_t stream) {
    (void)in_sizes; (void)n_in; (void)out_size; (void)ws_size;
    const float* X = (const float*)d_in[0];   // [8192,512]
    const int*   A = (const int*)  d_in[1];   // [8192,8192]
    const float* W = (const float*)d_in[2];   // [512,64]
    const float* a = (const float*)d_in[3];   // [128,1]

    // workspace layout (all 16B-aligned slices)
    float*     f1    = (float*)d_ws;                             // 8192
    float*     f2    = f1 + N_NODES;                             // 8192
    float*     Spart = f2 + N_NODES;                             // 4*8192*64
    float*     lpart = Spart + (size_t)NCHUNK * N_NODES * F_OUT; // 4*8192
    _Float16*  HT    = (_Float16*)(lpart + (size_t)NCHUNK * N_NODES); // 64*8192 f16

    gat_hidden_kernel<<<N_NODES, 64, 0, stream>>>(X, W, a, f1, f2, HT);
    gat_attn_kernel<<<(N_NODES / 16) * NCHUNK, 32, 0, stream>>>(A, f1, f2, HT, Spart, lpart);
    gat_final_kernel<<<(N_NODES * F_OUT + 255) / 256, 256, 0, stream>>>(Spart, lpart, (float*)d_out);
}
